// MOTIPCriterion_19937238188644
// MI455X (gfx1250) — compile-verified, matrix operand order checked by reading, not run on hardware
//
#include <hip/hip_runtime.h>
#include <hip/hip_bf16.h>

// MOTIP criterion for MI455X (gfx1250, wave32).
// Shapes (from setup_inputs): B=8,T=16 -> F=128 frames, N=300 preds,
// C=2 classes, M=32 targets, K=512 id classes.

#define NPRED 300
#define MTGT  32
#define NCLS  2
#define KID   512
#define HBLK  320               // hungarian block: one thread per column + pad

#define W_CLS_  2.0f
#define W_BBOX_ 5.0f
#define W_GIOU_ 2.0f
#define L_L1_   5.0f
#define L_GIOU_ 2.0f
#define L_ID_   1.0f

#define INFV 3.0e38f

typedef __attribute__((ext_vector_type(16))) _Float16 v16h;
typedef __attribute__((ext_vector_type(8)))  float    v8f;

// ---------------------------------------------------------------- GIoU helper
__device__ __forceinline__ float giou_cxcywh(const float p[4], const float t[4]) {
    const float ax0 = p[0] - 0.5f * p[2], ay0 = p[1] - 0.5f * p[3];
    const float ax1 = p[0] + 0.5f * p[2], ay1 = p[1] + 0.5f * p[3];
    const float bx0 = t[0] - 0.5f * t[2], by0 = t[1] - 0.5f * t[3];
    const float bx1 = t[0] + 0.5f * t[2], by1 = t[1] + 0.5f * t[3];
    const float area1 = (ax1 - ax0) * (ay1 - ay0);
    const float area2 = (bx1 - bx0) * (by1 - by0);
    const float iw = fmaxf(fminf(ax1, bx1) - fmaxf(ax0, bx0), 0.0f);
    const float ih = fmaxf(fminf(ay1, by1) - fmaxf(ay0, by0), 0.0f);
    const float inter = iw * ih;
    const float uni = area1 + area2 - inter;
    const float iou = inter / uni;
    const float cw = fmaxf(fmaxf(ax1, bx1) - fminf(ax0, bx0), 0.0f);
    const float ch = fmaxf(fmaxf(ay1, by1) - fminf(ay0, by0), 0.0f);
    const float areac = cw * ch;
    return iou - (areac - uni) / areac;
}

// ---------------------------------------------------------------- init
__global__ void motip_init_kernel(float* acc) {
    if (threadIdx.x < 3) acc[threadIdx.x] = 0.0f;
}

// ---------------------------------------------------------------- cost matrix
// cost[f][m][n] = W_CLS*(-prob[n,label[m]]) + W_BBOX*L1 + W_GIOU*(-giou)
__global__ __launch_bounds__(256) void motip_cost_kernel(
        const float* __restrict__ pl,      // [F,N,2]
        const float* __restrict__ pb,      // [F,N,4]
        const int*   __restrict__ labels,  // [F,M]
        const float* __restrict__ tb,      // [F,M,4]
        float* __restrict__ cost)          // [F,M,N]
{
    const int f = blockIdx.x;
    const int tid = threadIdx.x;

    __shared__ float s_pb[NPRED * 4];
    __shared__ float s_p0[NPRED];          // softmax prob of class 0
    __shared__ float s_tb[MTGT * 4];
    __shared__ int   s_lab[MTGT];

    for (int n = tid; n < NPRED; n += 256) {
        const float l0 = pl[((size_t)f * NPRED + n) * NCLS + 0];
        const float l1 = pl[((size_t)f * NPRED + n) * NCLS + 1];
        s_p0[n] = 1.0f / (1.0f + __expf(l1 - l0));
        const float4 b = *(const float4*)(pb + ((size_t)f * NPRED + n) * 4);
        s_pb[n * 4 + 0] = b.x; s_pb[n * 4 + 1] = b.y;
        s_pb[n * 4 + 2] = b.z; s_pb[n * 4 + 3] = b.w;
    }
    if (tid < MTGT) {
        s_lab[tid] = labels[f * MTGT + tid];
        const float4 b = *(const float4*)(tb + ((size_t)f * MTGT + tid) * 4);
        s_tb[tid * 4 + 0] = b.x; s_tb[tid * 4 + 1] = b.y;
        s_tb[tid * 4 + 2] = b.z; s_tb[tid * 4 + 3] = b.w;
    }
    __syncthreads();

    for (int e = tid; e < MTGT * NPRED; e += 256) {
        const int m = e / NPRED;
        const int n = e - m * NPRED;
        const float p0 = s_p0[n];
        const float plab = (s_lab[m] == 0) ? p0 : (1.0f - p0);
        float P[4] = {s_pb[n*4+0], s_pb[n*4+1], s_pb[n*4+2], s_pb[n*4+3]};
        float T[4] = {s_tb[m*4+0], s_tb[m*4+1], s_tb[m*4+2], s_tb[m*4+3]};
        const float cb = fabsf(P[0]-T[0]) + fabsf(P[1]-T[1]) +
                         fabsf(P[2]-T[2]) + fabsf(P[3]-T[3]);
        const float g = giou_cxcywh(P, T);
        cost[(size_t)f * MTGT * NPRED + e] =
            W_CLS_ * (-plab) + W_BBOX_ * cb + W_GIOU_ * (-g);
    }
}

// ---------------------------------------------------------------- hungarian
// Jonker-Volgenant shortest augmenting path, one block per frame.
// Thread tid owns column j = tid+1 (tid < 300): v[j], minv[j], used[j] live
// in registers; p/way/u live in LDS (serial augmentation on thread 0).
__global__ __launch_bounds__(HBLK) void motip_hungarian_kernel(
        const float* __restrict__ cost,    // [F,M,N] rows=targets
        int* __restrict__ pred_idx)        // [F,M]
{
    const int f = blockIdx.x;
    const int tid = threadIdx.x;
    const float* Cf = cost + (size_t)f * MTGT * NPRED;

    __shared__ int   sh_p[NPRED + 1];
    __shared__ int   sh_way[NPRED + 1];
    __shared__ float sh_u[MTGT + 1];
    __shared__ float red_val[512];
    __shared__ int   red_idx[512];
    __shared__ int   sh_j0;

    for (int j = tid; j <= NPRED; j += HBLK) { sh_p[j] = 0; sh_way[j] = 0; }
    for (int i = tid; i <= MTGT; i += HBLK) sh_u[i] = 0.0f;
    if (tid < 512 - HBLK) {                  // pad slots stay +inf forever
        red_val[HBLK + tid] = INFV;
        red_idx[HBLK + tid] = 1 << 29;
    }
    float my_v = 0.0f;
    const bool own = (tid < NPRED);
    __syncthreads();

    for (int i = 1; i <= MTGT; ++i) {
        if (tid == 0) { sh_p[0] = i; sh_j0 = 0; }
        float my_minv = INFV;
        bool my_used = false;
        bool used0 = false;                  // meaningful on tid 0 only
        __syncthreads();

        int j1 = 0;
        for (;;) {
            const int j0 = sh_j0;
            if (j0 == 0) { if (tid == 0) used0 = true; }
            else if (tid == j0 - 1) my_used = true;
            const int i0 = sh_p[j0];
            const float u_i0 = sh_u[i0];

            float mval = INFV;
            if (own && !my_used) {
                const float cur = Cf[(size_t)(i0 - 1) * NPRED + tid] - u_i0 - my_v;
                if (cur < my_minv) { my_minv = cur; sh_way[tid + 1] = j0; }
                mval = my_minv;
            }
            red_val[tid] = mval;
            red_idx[tid] = tid + 1;
            __syncthreads();
            for (int s = 256; s >= 1; s >>= 1) {
                if (tid < s) {
                    const float v2 = red_val[tid + s];
                    const int   x2 = red_idx[tid + s];
                    if (v2 < red_val[tid] ||
                        (v2 == red_val[tid] && x2 < red_idx[tid])) {
                        red_val[tid] = v2; red_idx[tid] = x2;
                    }
                }
                __syncthreads();
            }
            const float delta = red_val[0];
            j1 = red_idx[0];

            if (tid == 0 && used0) sh_u[sh_p[0]] += delta;   // column 0 (p[0]=i)
            if (own) {
                if (my_used) { sh_u[sh_p[tid + 1]] += delta; my_v -= delta; }
                else          my_minv -= delta;
            }
            if (tid == 0) sh_j0 = j1;
            __syncthreads();
            if (sh_p[j1] == 0) break;        // uniform across block
        }
        if (tid == 0) {                      // augment along `way`
            int j0 = j1;
            while (j0) { const int jn = sh_way[j0]; sh_p[j0] = sh_p[jn]; j0 = jn; }
        }
        __syncthreads();
    }
    if (own) {
        const int pj = sh_p[tid + 1];
        if (pj > 0) pred_idx[f * MTGT + pj - 1] = tid;  // col4row
    }
}

// ---------------------------------------------------------------- losses
// One block (2 waves) per frame. Wave w handles matched rows w*16..w*16+15
// for the ID loss. Log-sum-exp over K=512 is done on the matrix unit:
// D = A(16x32 f16 exp-values) x ones(32x16 f16) + C accumulates row sums in
// f32 across 16 chunks; f16 quantization residuals are compensated in f32.
// Chunk loops kept rolled (unroll 2) to stay under 256 VGPRs; __expf/__logf
// are TRANS32 ops that co-execute with the XDL WMMA pipeline.
__global__ __launch_bounds__(64) void motip_loss_kernel(
        const float* __restrict__ pb,      // [F,N,4]
        const float* __restrict__ il,      // [F,N,K]
        const float* __restrict__ tb,      // [F,M,4]
        const int*   __restrict__ tids,    // [F,M]
        const int*   __restrict__ pidx,    // [F,M]
        float* __restrict__ acc)           // [3]: l1_sum, giou_sum, nll_sum
{
    const int f    = blockIdx.x;
    const int tid  = threadIdx.x;
    const int wave = tid >> 5;
    const int lane = tid & 31;

    __shared__ float s_sums[MTGT];

    // ---- pair losses (wave 0, one matched pair per lane) ----
    float l1v = 0.0f, gv = 0.0f;
    if (wave == 0) {
        const int m  = lane;
        const int pi = pidx[f * MTGT + m];
        const float4 pB = *(const float4*)(pb + ((size_t)f * NPRED + pi) * 4);
        const float4 tB = *(const float4*)(tb + ((size_t)f * MTGT + m) * 4);
        float P[4] = {pB.x, pB.y, pB.z, pB.w};
        float T[4] = {tB.x, tB.y, tB.z, tB.w};
        l1v = fabsf(P[0]-T[0]) + fabsf(P[1]-T[1]) +
              fabsf(P[2]-T[2]) + fabsf(P[3]-T[3]);
        gv = giou_cxcywh(P, T);
    }
    #pragma unroll
    for (int s = 16; s >= 1; s >>= 1) {
        l1v += __shfl_xor(l1v, s);
        gv  += __shfl_xor(gv, s);
    }
    if (tid == 0) { atomicAdd(&acc[0], l1v); atomicAdd(&acc[1], gv); }

    // ---- ID loss: WMMA-based log-sum-exp over K=512 ----
    const int rowIdx = lane & 15;            // matrix row M this lane feeds
    const int half   = (lane >> 4) & 1;      // 0: K {0-7,16-23}; 1: K {8-15,24-31}
    const int off    = half * 8;
    const int m0     = wave * 16;
    const int pr     = pidx[f * MTGT + m0 + rowIdx];
    const float* R   = il + ((size_t)f * NPRED + pr) * KID;

    // pass 1: row max (each lane covers half the row; pairs combine via xor16)
    float mx = -INFV;
    #pragma unroll 2
    for (int c = 0; c < 16; ++c) {
        const int base = c * 32;
        const float4 a0 = *(const float4*)(R + base + off);
        const float4 a1 = *(const float4*)(R + base + off + 4);
        const float4 b0 = *(const float4*)(R + base + 16 + off);
        const float4 b1 = *(const float4*)(R + base + 16 + off + 4);
        mx = fmaxf(mx, fmaxf(fmaxf(fmaxf(a0.x, a0.y), fmaxf(a0.z, a0.w)),
                             fmaxf(fmaxf(a1.x, a1.y), fmaxf(a1.z, a1.w))));
        mx = fmaxf(mx, fmaxf(fmaxf(fmaxf(b0.x, b0.y), fmaxf(b0.z, b0.w)),
                             fmaxf(fmaxf(b1.x, b1.y), fmaxf(b1.z, b1.w))));
    }
    mx = fmaxf(mx, __shfl_xor(mx, 16));

    // pass 2: exp, f16 pack in ISA A-layout, WMMA ones-reduction + residual
    v16h onesv;
    #pragma unroll
    for (int h = 0; h < 16; ++h) onesv[h] = (_Float16)1.0f;
    v8f cacc = {};
    float losum = 0.0f;
    #pragma unroll 2
    for (int c = 0; c < 16; ++c) {
        const int base = c * 32;
        const float4 a0 = *(const float4*)(R + base + off);
        const float4 a1 = *(const float4*)(R + base + off + 4);
        const float4 b0 = *(const float4*)(R + base + 16 + off);
        const float4 b1 = *(const float4*)(R + base + 16 + off + 4);
        const float xs[16] = {a0.x, a0.y, a0.z, a0.w, a1.x, a1.y, a1.z, a1.w,
                              b0.x, b0.y, b0.z, b0.w, b1.x, b1.y, b1.z, b1.w};
        v16h afrag;
        #pragma unroll
        for (int h = 0; h < 16; ++h) {
            const float e = __expf(xs[h] - mx);   // v_exp_f32 (TRANS32)
            const _Float16 eh = (_Float16)e;
            losum += e - (float)eh;               // f16 quantization residual
            afrag[h] = eh;
        }
        // v_wmma_f32_16x16x32_f16: every column of D = row sums of A
        cacc = __builtin_amdgcn_wmma_f32_16x16x32_f16(
            false, afrag, false, onesv, (short)0, cacc, false, false);
    }
    losum += __shfl_xor(losum, 16);          // full-row residual

    // unpack C/D layout: VGPR v -> M=v (lanes 0-15) / M=v+8 (lanes 16-31)
    if (lane == 0) {
        #pragma unroll
        for (int v = 0; v < 8; ++v) s_sums[m0 + v] = cacc[v];
    }
    if (lane == 16) {
        #pragma unroll
        for (int v = 0; v < 8; ++v) s_sums[m0 + 8 + v] = cacc[v];
    }
    __syncthreads();

    float nll = 0.0f;
    if (lane < 16) {
        const int m = m0 + lane;             // rowIdx == lane here
        const float s = s_sums[m] + losum;
        const int   tk = tids[f * MTGT + m];
        const float xt = R[tk];
        nll = __logf(s) + mx - xt;           // -(xt - mx - log(sumexp))
    }
    #pragma unroll
    for (int s = 16; s >= 1; s >>= 1) nll += __shfl_xor(nll, s);
    if (lane == 0) atomicAdd(&acc[2], nll);
}

// ---------------------------------------------------------------- finalize
__global__ void motip_finalize_kernel(const float* __restrict__ acc,
                                      float* __restrict__ out, int F) {
    if (threadIdx.x == 0 && blockIdx.x == 0) {
        const float inv_pairs = 1.0f / (float)(F * MTGT);
        const float l1  = acc[0] * inv_pairs * 0.25f;   // mean over F*M*4
        const float gi  = 1.0f - acc[1] * inv_pairs;
        const float idl = acc[2] * inv_pairs;
        out[0] = L_L1_ * l1 + L_GIOU_ * gi + L_ID_ * idl;  // loss_cls == 0
        out[1] = 0.0f;
        out[2] = l1;
        out[3] = gi;
        out[4] = idl;
    }
}

// ---------------------------------------------------------------- launch
extern "C" void kernel_launch(void* const* d_in, const int* in_sizes, int n_in,
                              void* d_out, int out_size, void* d_ws, size_t ws_size,
                              hipStream_t stream) {
    (void)n_in; (void)out_size; (void)ws_size;
    const float* pl     = (const float*)d_in[0];   // pred_logits  [F,N,2]
    const float* pb     = (const float*)d_in[1];   // pred_boxes   [F,N,4]
    const float* il     = (const float*)d_in[2];   // id_logits    [F,N,512]
    const int*   labels = (const int*)  d_in[3];   // target_labels[F,M]
    const float* tb     = (const float*)d_in[4];   // target_boxes [F,M,4]
    const int*   tids   = (const int*)  d_in[5];   // target_ids   [F,M]
    const int F = in_sizes[5] / MTGT;              // 128

    char* ws = (char*)d_ws;
    float* cost = (float*)ws;
    size_t off  = (size_t)F * MTGT * NPRED * sizeof(float);
    int*   pidx = (int*)(ws + off);
    off += (size_t)F * MTGT * sizeof(int);
    float* acc  = (float*)(ws + off);

    motip_init_kernel<<<1, 32, 0, stream>>>(acc);
    motip_cost_kernel<<<F, 256, 0, stream>>>(pl, pb, labels, tb, cost);
    motip_hungarian_kernel<<<F, HBLK, 0, stream>>>(cost, pidx);
    motip_loss_kernel<<<F, 64, 0, stream>>>(pb, il, tb, tids, pidx, acc);
    motip_finalize_kernel<<<1, 32, 0, stream>>>(acc, (float*)d_out, F);
}
